// RGDTEncoder_19310172963038
// MI455X (gfx1250) — compile-verified, matrix operand order checked by reading, not run on hardware
//
#include <hip/hip_runtime.h>
#include <hip/hip_bf16.h>
#include <math.h>

#define NN 50000          // nodes
#define NE 800000         // edges
#define D  128            // dim
#define H  8              // heads
#define HD 16             // head dim
#define FFN 512
#define B  512            // batch
#define NREL 500
#define ALPHA 0.15f
#define SLOPE 0.2f

#define CDIV(a,b) (((a)+(b)-1)/(b))

typedef __attribute__((ext_vector_type(16))) __bf16 v16bf;
typedef __attribute__((ext_vector_type(8)))  __bf16 v8bf;
typedef __attribute__((ext_vector_type(8)))  float  v8f;

// ---------- ordered-float helpers for atomic max on f32 ----------
__device__ __forceinline__ unsigned f2ord(float f) {
    unsigned u = __float_as_uint(f);
    return (u & 0x80000000u) ? ~u : (u | 0x80000000u);
}
__device__ __forceinline__ float ord2f(unsigned u) {
    u = (u & 0x80000000u) ? (u & 0x7fffffffu) : ~u;
    return __uint_as_float(u);
}

// ---------- h = ent_tab[ent_ids] + pos_tab[arw_pos] ----------
__global__ void embed_kernel(const int* __restrict__ ent_ids,
                             const int* __restrict__ arw_pos,
                             const float* __restrict__ ent_tab,
                             const float* __restrict__ pos_tab,
                             float* __restrict__ h) {
    int idx = blockIdx.x * blockDim.x + threadIdx.x;
    if (idx >= NN * D) return;
    int n = idx >> 7, d = idx & 127;
    h[idx] = ent_tab[(size_t)ent_ids[n] * D + d] + pos_tab[(size_t)arw_pos[n] * D + d];
}

// ---------- one-time weight convert + transpose: out[n*K+k] = (bf16)in[k*N+n] ----------
__global__ void cvt_bf16_t_kernel(const float* __restrict__ in,
                                  __bf16* __restrict__ out, int Kd, int Nd) {
    int idx = blockIdx.x * blockDim.x + threadIdx.x;
    if (idx >= Kd * Nd) return;
    int n = idx / Kd, k = idx % Kd;           // output-coalesced order
    out[idx] = (__bf16)in[(size_t)k * Nd + n];
}

// ---------- fold: out[d*H+h] = sum_k W[d*D + h*16+k] * avec[h*16+k] ----------
__global__ void fold_kernel(const float* __restrict__ W,
                            const float* __restrict__ avec,
                            float* __restrict__ out) {
    int idx = blockIdx.x * blockDim.x + threadIdx.x;
    if (idx >= D * H) return;
    int d = idx >> 3, hh = idx & 7;
    float s = 0.f;
    #pragma unroll
    for (int k = 0; k < HD; ++k) s += W[(size_t)d * D + hh * HD + k] * avec[hh * HD + k];
    out[idx] = s;
}

// ---------- per-relation scores: relsc[r*H+h] = rel_tab[r,:] . fold[:,h] ----------
__global__ void relscore_kernel(const float* __restrict__ rel_tab,
                                const float* __restrict__ fold,
                                float* __restrict__ relsc) {
    int idx = blockIdx.x * blockDim.x + threadIdx.x;
    if (idx >= NREL * H) return;
    int r = idx >> 3, hh = idx & 7;
    float s = 0.f;
    for (int d = 0; d < D; ++d) s += rel_tab[(size_t)r * D + d] * fold[d * H + hh];
    relsc[idx] = s;
}

// ---------- double-buffered LDS-staged bf16 WMMA GEMM ----------
// C[M,N] = A[M,K](f32) @ B (pre-converted, pre-transposed bf16: Bt[n*K+k])
// block: 256 thr = 8 waves; macro-tile 64x64; each wave: 16 rows x two 16x16 tiles
// Software pipelined: prefetch chunk k+1 to regs while WMMAs consume LDS chunk k.
#define BM 64
#define BN 64
#define BK 32

template<bool RELU, bool HASBIAS>
__global__ void __launch_bounds__(256)
gemm_tile_kernel(const float* __restrict__ A, const __bf16* __restrict__ Bt,
                 const float* __restrict__ bias, float* __restrict__ C,
                 int M, int N, int K) {
    __shared__ __bf16 Asm[2][BM][BK];   // [buf][m][k]
    __shared__ __bf16 Bsm[2][BN][BK];   // [buf][n][k]

    int nblocks = N / BN;
    int m0 = (blockIdx.x / nblocks) * BM;
    int n0 = (blockIdx.x % nblocks) * BN;

    int tid  = threadIdx.x;
    int wave = tid >> 5;         // 0..7
    int lane = tid & 31;
    int wm = wave >> 1;          // 0..3  M sub-tile
    int wn = wave & 1;           // 0..1  N sub-tile pair

    int frow = lane & 15;
    int fkb  = (lane >> 4) << 3;

    int sr = tid >> 2;           // staging row 0..63
    int sc = (tid & 3) << 3;     // staging col 0,8,16,24

    int gm = m0 + sr; if (gm >= M) gm = M - 1;   // clamp tail once
    const float*  abase = A  + (size_t)gm * K + sc;
    const __bf16* bbase = Bt + (size_t)(n0 + sr) * K + sc;

    v8f acc0 = {};
    v8f acc1 = {};

    float4 p0, p1, bq;

    // prologue: load + stage chunk 0 into buffer 0
    p0 = *(const float4*)(abase);
    p1 = *(const float4*)(abase + 4);
    bq = *(const float4*)(bbase);
    {
        v8bf pk;
        pk[0] = (__bf16)p0.x; pk[1] = (__bf16)p0.y;
        pk[2] = (__bf16)p0.z; pk[3] = (__bf16)p0.w;
        pk[4] = (__bf16)p1.x; pk[5] = (__bf16)p1.y;
        pk[6] = (__bf16)p1.z; pk[7] = (__bf16)p1.w;
        *(v8bf*)&Asm[0][sr][sc] = pk;
        *(float4*)&Bsm[0][sr][sc] = bq;
    }
    __syncthreads();

    int buf = 0;
    for (int k0 = BK; k0 < K; k0 += BK) {
        // prefetch next chunk into registers (overlaps with WMMAs below)
        p0 = *(const float4*)(abase + k0);
        p1 = *(const float4*)(abase + k0 + 4);
        bq = *(const float4*)(bbase + k0);

        // compute on current buffer
        {
            v16bf af, bf0, bf1;
            #pragma unroll
            for (int i = 0; i < 8; ++i) {
                af[i]      = Asm[buf][wm * 16 + frow][fkb + i];
                af[i + 8]  = Asm[buf][wm * 16 + frow][16 + fkb + i];
                bf0[i]     = Bsm[buf][wn * 32 + frow][fkb + i];
                bf0[i + 8] = Bsm[buf][wn * 32 + frow][16 + fkb + i];
                bf1[i]     = Bsm[buf][wn * 32 + 16 + frow][fkb + i];
                bf1[i + 8] = Bsm[buf][wn * 32 + 16 + frow][16 + fkb + i];
            }
            acc0 = __builtin_amdgcn_wmma_f32_16x16x32_bf16(false, af, false, bf0,
                                                           (short)0, acc0, false, false);
            acc1 = __builtin_amdgcn_wmma_f32_16x16x32_bf16(false, af, false, bf1,
                                                           (short)0, acc1, false, false);
        }

        // stage prefetched chunk into alternate buffer
        buf ^= 1;
        {
            v8bf pk;
            pk[0] = (__bf16)p0.x; pk[1] = (__bf16)p0.y;
            pk[2] = (__bf16)p0.z; pk[3] = (__bf16)p0.w;
            pk[4] = (__bf16)p1.x; pk[5] = (__bf16)p1.y;
            pk[6] = (__bf16)p1.z; pk[7] = (__bf16)p1.w;
            *(v8bf*)&Asm[buf][sr][sc] = pk;
            *(float4*)&Bsm[buf][sr][sc] = bq;
        }
        __syncthreads();
    }

    // epilogue compute on final buffer
    {
        v16bf af, bf0, bf1;
        #pragma unroll
        for (int i = 0; i < 8; ++i) {
            af[i]      = Asm[buf][wm * 16 + frow][fkb + i];
            af[i + 8]  = Asm[buf][wm * 16 + frow][16 + fkb + i];
            bf0[i]     = Bsm[buf][wn * 32 + frow][fkb + i];
            bf0[i + 8] = Bsm[buf][wn * 32 + frow][16 + fkb + i];
            bf1[i]     = Bsm[buf][wn * 32 + 16 + frow][fkb + i];
            bf1[i + 8] = Bsm[buf][wn * 32 + 16 + frow][16 + fkb + i];
        }
        acc0 = __builtin_amdgcn_wmma_f32_16x16x32_bf16(false, af, false, bf0,
                                                       (short)0, acc0, false, false);
        acc1 = __builtin_amdgcn_wmma_f32_16x16x32_bf16(false, af, false, bf1,
                                                       (short)0, acc1, false, false);
    }

    int nl = lane & 15;
    int mb = m0 + wm * 16 + ((lane >> 4) << 3);
    int nA = n0 + wn * 32 + nl;
    int nB = nA + 16;
    float bvA = 0.f, bvB = 0.f;
    if (HASBIAS) { bvA = bias[nA]; bvB = bias[nB]; }

    if (m0 + BM <= M) {
        // fast path: block fully inside M, no predication
        #pragma unroll
        for (int v = 0; v < 8; ++v) {
            float x0 = acc0[v];
            float x1 = acc1[v];
            if (HASBIAS) { x0 += bvA; x1 += bvB; }
            if (RELU)    { x0 = fmaxf(x0, 0.f); x1 = fmaxf(x1, 0.f); }
            C[(size_t)(mb + v) * N + nA] = x0;
            C[(size_t)(mb + v) * N + nB] = x1;
        }
    } else {
        #pragma unroll
        for (int v = 0; v < 8; ++v) {
            int m = mb + v;
            if (m < M) {
                float x0 = acc0[v];
                float x1 = acc1[v];
                if (HASBIAS) { x0 += bvA; x1 += bvB; }
                if (RELU)    { x0 = fmaxf(x0, 0.f); x1 = fmaxf(x1, 0.f); }
                C[(size_t)m * N + nA] = x0;
                C[(size_t)m * N + nB] = x1;
            }
        }
    }
}

// ---------- head scores: s_src/s_dst [NN,H] from f [NN,D] ----------
__global__ void head_score_kernel(const float* __restrict__ f,
                                  const float* __restrict__ ah,
                                  const float* __restrict__ at,
                                  float* __restrict__ ssrc,
                                  float* __restrict__ sdst) {
    int idx = blockIdx.x * blockDim.x + threadIdx.x;
    if (idx >= NN * H) return;
    int n = idx >> 3, hh = idx & 7;
    const float* fr = f + (size_t)n * D + hh * HD;
    float s0 = 0.f, s1 = 0.f;
    #pragma unroll
    for (int k = 0; k < HD; ++k) {
        float v = fr[k];
        s0 += v * ah[hh * HD + k];
        s1 += v * at[hh * HD + k];
    }
    ssrc[idx] = s0;
    sdst[idx] = s1;
}

__global__ void init_max_kernel(unsigned* __restrict__ mord) {
    int idx = blockIdx.x * blockDim.x + threadIdx.x;
    if (idx >= NN * H) return;
    mord[idx] = f2ord(-INFINITY);
}

// ---------- edge logits + leaky relu + segment max : one thread per edge ----------
__global__ void edge_logit_kernel(const int* __restrict__ src,
                                  const int* __restrict__ dst,
                                  const int* __restrict__ rel_ids,
                                  const float* __restrict__ ssrc,
                                  const float* __restrict__ sdst,
                                  const float* __restrict__ relsc,
                                  float* __restrict__ ebuf,
                                  unsigned* __restrict__ mord) {
    int e = blockIdx.x * blockDim.x + threadIdx.x;
    if (e >= NE) return;
    int s = src[e], t = dst[e];
    const float* ps = ssrc + (size_t)s * H;
    const float* pt = sdst + (size_t)t * H;
    const float* pr = relsc ? relsc + (size_t)rel_ids[e] * H : nullptr;
    #pragma unroll
    for (int hh = 0; hh < H; ++hh) {
        float v = ps[hh] + pt[hh];
        if (pr) v += pr[hh];
        v = (v >= 0.f) ? v : SLOPE * v;
        ebuf[(size_t)e * H + hh] = v;
        atomicMax(&mord[t * H + hh], f2ord(v));
    }
}

// ---------- decode max (non-finite -> 0) and zero denominators ----------
__global__ void resolve_max_kernel(unsigned* __restrict__ mord,
                                   float* __restrict__ mout,
                                   float* __restrict__ den) {
    int idx = blockIdx.x * blockDim.x + threadIdx.x;
    if (idx >= NN * H) return;
    float m = ord2f(mord[idx]);
    if (!isfinite(m)) m = 0.f;
    mout[idx] = m;
    den[idx] = 0.f;
}

// ---------- exp(e - m[dst]) + segment sum : one thread per edge ----------
__global__ void edge_exp_kernel(const int* __restrict__ dst,
                                float* __restrict__ ebuf,
                                const float* __restrict__ m,
                                float* __restrict__ den) {
    int e = blockIdx.x * blockDim.x + threadIdx.x;
    if (e >= NE) return;
    int t = dst[e];
    #pragma unroll
    for (int hh = 0; hh < H; ++hh) {
        float v = expf(ebuf[(size_t)e * H + hh] - m[t * H + hh]);
        ebuf[(size_t)e * H + hh] = v;
        atomicAdd(&den[t * H + hh], v);
    }
}

// ---------- att = ex / (den[dst] + eps) : one thread per edge ----------
__global__ void edge_att_kernel(const int* __restrict__ dst,
                                float* __restrict__ ebuf,
                                const float* __restrict__ den) {
    int e = blockIdx.x * blockDim.x + threadIdx.x;
    if (e >= NE) return;
    int t = dst[e];
    #pragma unroll
    for (int hh = 0; hh < H; ++hh)
        ebuf[(size_t)e * H + hh] /= (den[t * H + hh] + 1e-16f);
}

__global__ void zero_kernel(float* __restrict__ p, int count4) {
    int idx = blockIdx.x * blockDim.x + threadIdx.x;
    if (idx < count4) ((float4*)p)[idx] = make_float4(0.f, 0.f, 0.f, 0.f);
}

// ---------- hop scatter: one wave per edge, float4 per lane ----------
__global__ void hop_scatter_kernel(const int* __restrict__ src,
                                   const int* __restrict__ dst,
                                   const float* __restrict__ att,
                                   const float* __restrict__ feat,
                                   float* __restrict__ agg) {
    int idx = blockIdx.x * blockDim.x + threadIdx.x;
    int e = idx >> 5;
    if (e >= NE) return;
    int lane = idx & 31;
    int d0 = lane << 2;                          // 0..124
    int s = src[e], t = dst[e];
    float a = att[(size_t)e * H + (d0 >> 4)];    // head = d0/16
    float4 fv = *(const float4*)(feat + (size_t)s * D + d0);
    float* ag = agg + (size_t)t * D + d0;
    atomicAdd(ag + 0, fv.x * a);
    atomicAdd(ag + 1, fv.y * a);
    atomicAdd(ag + 2, fv.z * a);
    atomicAdd(ag + 3, fv.w * a);
}

// ---------- feat = alpha*f0 + (1-alpha)*agg ; agg = 0 ----------
__global__ void hop_combine_kernel(const float* __restrict__ f0,
                                   float* __restrict__ agg,
                                   float* __restrict__ feat) {
    int idx = blockIdx.x * blockDim.x + threadIdx.x;
    if (idx >= NN * D / 4) return;
    float4 f = ((const float4*)f0)[idx];
    float4 g = ((float4*)agg)[idx];
    float4 r;
    r.x = ALPHA * f.x + (1.f - ALPHA) * g.x;
    r.y = ALPHA * f.y + (1.f - ALPHA) * g.y;
    r.z = ALPHA * f.z + (1.f - ALPHA) * g.z;
    r.w = ALPHA * f.w + (1.f - ALPHA) * g.w;
    ((float4*)feat)[idx] = r;
    ((float4*)agg)[idx] = make_float4(0.f, 0.f, 0.f, 0.f);
}

// ---------- out = LayerNorm(x + res) * g + b : one wave32 per row ----------
__global__ void add_ln_kernel(const float* __restrict__ x,
                              const float* __restrict__ res,
                              const float* __restrict__ g,
                              const float* __restrict__ b,
                              float* __restrict__ out) {
    int gtid = blockIdx.x * blockDim.x + threadIdx.x;
    int row = gtid >> 5;
    int lane = gtid & 31;
    if (row >= NN) return;
    const float* xr = x + (size_t)row * D;
    const float* rr = res + (size_t)row * D;
    float v[4], s = 0.f, s2 = 0.f;
    #pragma unroll
    for (int i = 0; i < 4; ++i) {
        int d = lane + 32 * i;
        v[i] = xr[d] + rr[d];
        s += v[i];
        s2 += v[i] * v[i];
    }
    #pragma unroll
    for (int off = 16; off > 0; off >>= 1) {
        s  += __shfl_xor(s, off, 32);
        s2 += __shfl_xor(s2, off, 32);
    }
    float mean = s * (1.f / D);
    float var  = s2 * (1.f / D) - mean * mean;
    float inv  = rsqrtf(var + 1e-5f);
    #pragma unroll
    for (int i = 0; i < 4; ++i) {
        int d = lane + 32 * i;
        out[(size_t)row * D + d] = (v[i] - mean) * inv * g[d] + b[d];
    }
}

// ---------- final batch gather ----------
__global__ void gather_kernel(const float* __restrict__ h,
                              const int* __restrict__ ids,
                              float* __restrict__ out) {
    int idx = blockIdx.x * blockDim.x + threadIdx.x;
    if (idx >= B * D) return;
    int bi = idx >> 7, d = idx & 127;
    out[idx] = h[(size_t)ids[bi] * D + d];
}

extern "C" void kernel_launch(void* const* d_in, const int* in_sizes, int n_in,
                              void* d_out, int out_size, void* d_ws, size_t ws_size,
                              hipStream_t stream) {
    const int*   ent_ids  = (const int*)d_in[0];
    const int*   rel_ids  = (const int*)d_in[1];
    const int*   arw_pos  = (const int*)d_in[2];
    const int*   src      = (const int*)d_in[3];
    const int*   dst      = (const int*)d_in[4];
    const int*   batchids = (const int*)d_in[5];
    const float* ent_tab  = (const float*)d_in[6];
    const float* rel_tab  = (const float*)d_in[7];
    const float* pos_tab  = (const float*)d_in[8];

    const float* l0_Wh = (const float*)d_in[9];
    const float* l0_Wr = (const float*)d_in[10];
    const float* l0_ah = (const float*)d_in[11];
    const float* l0_at = (const float*)d_in[12];
    const float* l0_ar = (const float*)d_in[13];
    const float* l0_g1 = (const float*)d_in[14];
    const float* l0_b1 = (const float*)d_in[15];
    const float* l0_w1 = (const float*)d_in[16];
    const float* l0_c1 = (const float*)d_in[17];
    const float* l0_w2 = (const float*)d_in[18];
    const float* l0_c2 = (const float*)d_in[19];
    const float* l0_g2 = (const float*)d_in[20];
    const float* l0_b2 = (const float*)d_in[21];
    const float* l1_Wh = (const float*)d_in[22];
    const float* l1_ah = (const float*)d_in[23];
    const float* l1_at = (const float*)d_in[24];
    const float* l1_g1 = (const float*)d_in[25];
    const float* l1_b1 = (const float*)d_in[26];
    const float* l1_w1 = (const float*)d_in[27];
    const float* l1_c1 = (const float*)d_in[28];
    const float* l1_w2 = (const float*)d_in[29];
    const float* l1_c2 = (const float*)d_in[30];
    const float* l1_g2 = (const float*)d_in[31];
    const float* l1_b2 = (const float*)d_in[32];

    // ---- workspace layout ----
    float* W = (float*)d_ws;
    size_t off = 0;
    float* h_buf  = W + off; off += (size_t)NN * D;
    float* f0_buf = W + off; off += (size_t)NN * D;   // proj / feat0 / ffn_out
    float* feat   = W + off; off += (size_t)NN * D;
    float* agg    = W + off; off += (size_t)NN * D;
    float* o_buf  = W + off; off += (size_t)NN * D;
    float* ssrc   = W + off; off += (size_t)NN * H;
    float* sdst   = W + off; off += (size_t)NN * H;
    float* mmax   = W + off; off += (size_t)NN * H;
    float* den    = W + off; off += (size_t)NN * H;
    float* relsc  = W + off; off += (size_t)512 * H;
    float* wfold  = W + off; off += (size_t)D * H;
    float* ebuf   = W + off; off += (size_t)NE * H;
    float* mid    = W + off; off += (size_t)NN * FFN;
    // bf16 transposed weight copies (converted once per call)
    __bf16* bw = (__bf16*)(W + off);
    size_t boff = 0;
    __bf16* l0Wh_b = bw + boff; boff += (size_t)D * D;     // [n=128][k=128]
    __bf16* l0w1_b = bw + boff; boff += (size_t)D * FFN;   // [n=512][k=128]
    __bf16* l0w2_b = bw + boff; boff += (size_t)FFN * D;   // [n=128][k=512]
    __bf16* l1Wh_b = bw + boff; boff += (size_t)D * D;
    __bf16* l1w1_b = bw + boff; boff += (size_t)D * FFN;
    __bf16* l1w2_b = bw + boff; boff += (size_t)FFN * D;

    const int BLK = 256;
    unsigned* mord = (unsigned*)mmax;

    // ---- embeddings + one-time weight convert/transpose ----
    embed_kernel<<<CDIV(NN * D, BLK), BLK, 0, stream>>>(ent_ids, arw_pos, ent_tab, pos_tab, h_buf);
    cvt_bf16_t_kernel<<<CDIV(D * D, BLK), BLK, 0, stream>>>(l0_Wh, l0Wh_b, D, D);
    cvt_bf16_t_kernel<<<CDIV(D * FFN, BLK), BLK, 0, stream>>>(l0_w1, l0w1_b, D, FFN);
    cvt_bf16_t_kernel<<<CDIV(FFN * D, BLK), BLK, 0, stream>>>(l0_w2, l0w2_b, FFN, D);
    cvt_bf16_t_kernel<<<CDIV(D * D, BLK), BLK, 0, stream>>>(l1_Wh, l1Wh_b, D, D);
    cvt_bf16_t_kernel<<<CDIV(D * FFN, BLK), BLK, 0, stream>>>(l1_w1, l1w1_b, D, FFN);
    cvt_bf16_t_kernel<<<CDIV(FFN * D, BLK), BLK, 0, stream>>>(l1_w2, l1w2_b, FFN, D);

    // ---- per-relation attention scores (rel_score depends only on rel id) ----
    fold_kernel<<<CDIV(D * H, BLK), BLK, 0, stream>>>(l0_Wr, l0_ar, wfold);
    relscore_kernel<<<CDIV(NREL * H, BLK), BLK, 0, stream>>>(rel_tab, wfold, relsc);

    auto run_layer = [&](const __bf16* Wh_b, const float* ah, const float* at,
                         const float* g1, const float* b1,
                         const __bf16* w1_b, const float* c1,
                         const __bf16* w2_b, const float* c2,
                         const float* g2, const float* b2,
                         const float* rel_score_tab) {
        // f0 = h @ Wh  [NN,128]x[128,128]
        gemm_tile_kernel<false, false><<<CDIV(NN, BM) * (D / BN), BLK, 0, stream>>>(
            h_buf, Wh_b, nullptr, f0_buf, NN, D, D);
        head_score_kernel<<<CDIV(NN * H, BLK), BLK, 0, stream>>>(f0_buf, ah, at, ssrc, sdst);

        init_max_kernel<<<CDIV(NN * H, BLK), BLK, 0, stream>>>(mord);
        edge_logit_kernel<<<CDIV(NE, BLK), BLK, 0, stream>>>(
            src, dst, rel_ids, ssrc, sdst, rel_score_tab, ebuf, mord);
        resolve_max_kernel<<<CDIV(NN * H, BLK), BLK, 0, stream>>>(mord, mmax, den);
        edge_exp_kernel<<<CDIV(NE, BLK), BLK, 0, stream>>>(dst, ebuf, mmax, den);
        edge_att_kernel<<<CDIV(NE, BLK), BLK, 0, stream>>>(dst, ebuf, den);

        // ---- PPR hops ----
        zero_kernel<<<CDIV(NN * D / 4, BLK), BLK, 0, stream>>>(agg, NN * D / 4);
        const float* cur = f0_buf;
        for (int hop = 0; hop < 3; ++hop) {
            hop_scatter_kernel<<<CDIV(NE * 32, BLK), BLK, 0, stream>>>(src, dst, ebuf, cur, agg);
            hop_combine_kernel<<<CDIV(NN * D / 4, BLK), BLK, 0, stream>>>(f0_buf, agg, feat);
            cur = feat;
        }

        // o = LN(feat + h)
        add_ln_kernel<<<CDIV(NN * 32, BLK), BLK, 0, stream>>>(feat, h_buf, g1, b1, o_buf);

        // FFN
        gemm_tile_kernel<true, true><<<CDIV(NN, BM) * (FFN / BN), BLK, 0, stream>>>(
            o_buf, w1_b, c1, mid, NN, FFN, D);
        gemm_tile_kernel<false, true><<<CDIV(NN, BM) * (D / BN), BLK, 0, stream>>>(
            mid, w2_b, c2, f0_buf, NN, D, FFN);

        // h = LN(o + ffn_out)
        add_ln_kernel<<<CDIV(NN * 32, BLK), BLK, 0, stream>>>(f0_buf, o_buf, g2, b2, h_buf);
    };

    run_layer(l0Wh_b, l0_ah, l0_at, l0_g1, l0_b1, l0w1_b, l0_c1, l0w2_b, l0_c2, l0_g2, l0_b2, relsc);
    run_layer(l1Wh_b, l1_ah, l1_at, l1_g1, l1_b1, l1w1_b, l1_c1, l1w2_b, l1_c2, l1_g2, l1_b2, nullptr);

    gather_kernel<<<CDIV(B * D, BLK), BLK, 0, stream>>>(h_buf, batchids, (float*)d_out);
    (void)in_sizes; (void)n_in; (void)out_size; (void)ws_size;
}
